// NodeRouter_83124797047150
// MI455X (gfx1250) — matrix operand order repaired and checked by reading, not compiled, not run on hardware
//
#include <hip/hip_runtime.h>
#include <hip/hip_bf16.h>

#define B_   8
#define L_   2048
#define D_   1024
#define NN   64
#define NH   8
#define NS   (NN * NH)   // 512 gate slots
#define KSEL 2

typedef __attribute__((ext_vector_type(16))) __bf16 v16bf;
typedef __attribute__((ext_vector_type(8)))  __bf16 v8bf;
typedef __attribute__((ext_vector_type(8)))  float  v8f;

// ---------------------------------------------------------------------------
// Kernel 0: one-time W f32 -> bf16 (RNE). 1 MB result stays L2-resident.
// ---------------------------------------------------------------------------
__global__ __launch_bounds__(256)
void cvt_w_bf16(const float4* __restrict__ W4, ushort4* __restrict__ Wb4) {
    int i = blockIdx.x * 256 + threadIdx.x;     // exactly NS*D_/4 threads
    float4 v = W4[i];
    ushort4 o;
    o.x = __builtin_bit_cast(unsigned short, (__bf16)v.x);
    o.y = __builtin_bit_cast(unsigned short, (__bf16)v.y);
    o.z = __builtin_bit_cast(unsigned short, (__bf16)v.z);
    o.w = __builtin_bit_cast(unsigned short, (__bf16)v.w);
    Wb4[i] = o;
}

// ---------------------------------------------------------------------------
// top-2 merge helper: merge sorted pair (w1>=w2) into sorted pair (v1>=v2)
// ---------------------------------------------------------------------------
__device__ __forceinline__
void top2_merge(float& v1, int& i1, float& v2, int& i2,
                float w1, int j1, float w2, int j2) {
    if (w1 > v1) {
        if (v1 >= w2) { v2 = v1; i2 = i1; }
        else          { v2 = w2; i2 = j2; }
        v1 = w1; i1 = j1;
    } else if (w1 > v2) {
        v2 = w1; i2 = j1;
    }
}

// ---------------------------------------------------------------------------
// Kernel 1: gate logits via bf16 WMMA (f32 accumulate) + per-token top-2 +
//           exact f32 recompute of the two selected logits.
// Block = 512 threads (16 waves); block owns 16 tokens (one M-tile).
// Wave w computes N-tiles 2w and 2w+1 (two 16x16 C tiles).
// ---------------------------------------------------------------------------
#define LROW 517   // LDS row stride (floats) to avoid stride-512 bank conflicts

__global__ __launch_bounds__(512)
void gate_topk(const float*  __restrict__ x,
               const float*  __restrict__ W,      // f32, for exact recompute
               const __bf16* __restrict__ Wb,     // bf16 copy
               const float*  __restrict__ bias,
               float*        __restrict__ sel_val,
               int*          __restrict__ sel_idx) {
    __shared__ float lds[16 * LROW];

    const int tid  = threadIdx.x;
    const int wv   = tid >> 5;      // wave 0..15
    const int lane = tid & 31;
    const int lh   = lane & 15;
    const int hi   = lane >> 4;     // lane group (0: lanes 0-15, 1: 16-31)

    // A-matrix row (token) for this lane
    const int    mrow = blockIdx.x * 16 + lh;
    const float* xr   = x + (size_t)mrow * D_;

    // B-matrix rows (gate slots) for this wave's two N-tiles
    const int slot0 = (2 * wv)     * 16 + lh;
    const int slot1 = (2 * wv + 1) * 16 + lh;
    const __bf16* wr0 = Wb + (size_t)slot0 * D_;
    const __bf16* wr1 = Wb + (size_t)slot1 * D_;

    v8f acc0 = {};
    v8f acc1 = {};

    for (int k0 = 0; k0 < D_; k0 += 32) {
        // ---- A: 16 bf16 per lane. Layout => elements 0..7 = d[bd..bd+7],
        //      elements 8..15 = d[bd+16..bd+23], bd = k0 + hi*8.
        const int bd = k0 + hi * 8;
        float4 a0 = *(const float4*)(xr + bd);
        float4 a1 = *(const float4*)(xr + bd + 4);
        float4 a2 = *(const float4*)(xr + bd + 16);
        float4 a3 = *(const float4*)(xr + bd + 20);
        v16bf av;
        av[0]  = (__bf16)a0.x; av[1]  = (__bf16)a0.y;
        av[2]  = (__bf16)a0.z; av[3]  = (__bf16)a0.w;
        av[4]  = (__bf16)a1.x; av[5]  = (__bf16)a1.y;
        av[6]  = (__bf16)a1.z; av[7]  = (__bf16)a1.w;
        av[8]  = (__bf16)a2.x; av[9]  = (__bf16)a2.y;
        av[10] = (__bf16)a2.z; av[11] = (__bf16)a2.w;
        av[12] = (__bf16)a3.x; av[13] = (__bf16)a3.y;
        av[14] = (__bf16)a3.z; av[15] = (__bf16)a3.w;

        // ---- B: same k layout as A: d[wd..wd+7], d[wd+16..wd+23], wd = k0 + hi*8.
        const int wd = k0 + hi * 8;
        v8bf b0l = *(const v8bf*)(wr0 + wd);
        v8bf b0h = *(const v8bf*)(wr0 + wd + 16);
        v8bf b1l = *(const v8bf*)(wr1 + wd);
        v8bf b1h = *(const v8bf*)(wr1 + wd + 16);
        v16bf bv0, bv1;
        #pragma unroll
        for (int j = 0; j < 8; ++j) {
            bv0[j] = b0l[j]; bv0[8 + j] = b0h[j];
            bv1[j] = b1l[j]; bv1[8 + j] = b1h[j];
        }

        acc0 = __builtin_amdgcn_wmma_f32_16x16x32_bf16(
                   false, av, false, bv0, (short)0, acc0, false, false);
        acc1 = __builtin_amdgcn_wmma_f32_16x16x32_bf16(
                   false, av, false, bv1, (short)0, acc1, false, false);
    }

    // Add bias, spill logits to LDS: lds[token_local][slot]
    const float bz0 = bias[slot0];
    const float bz1 = bias[slot1];
    #pragma unroll
    for (int r = 0; r < 8; ++r) {
        const int row = hi * 8 + r;   // C layout: lanes16-31 hold M=8..15
        lds[row * LROW + slot0] = acc0[r] + bz0;
        lds[row * LROW + slot1] = acc1[r] + bz1;
    }
    __syncthreads();

    // ---- per-token top-2: wave wv scans the 512 logits of token wv
    float v1 = -1e30f, v2 = -1e30f;
    int   i1 = 0,      i2 = 0;
    const float* row = &lds[wv * LROW];
    for (int s = lane; s < NS; s += 32)
        top2_merge(v1, i1, v2, i2, row[s], s, -1e30f, -1);
    #pragma unroll
    for (int m = 16; m >= 1; m >>= 1) {
        float w1 = __shfl_xor(v1, m); int j1 = __shfl_xor(i1, m);
        float w2 = __shfl_xor(v2, m); int j2 = __shfl_xor(i2, m);
        top2_merge(v1, i1, v2, i2, w1, j1, w2, j2);
    }

    // ---- exact f32 recompute of the two selected logits (wave-parallel)
    const int    tok = blockIdx.x * 16 + wv;
    const float* xt  = x + (size_t)tok * D_;
    const float* w1r = W + (size_t)i1 * D_;
    const float* w2r = W + (size_t)i2 * D_;
    float d1 = 0.f, d2 = 0.f;
    for (int d = lane; d < D_; d += 32) {
        const float xv = xt[d];
        d1 = fmaf(xv, w1r[d], d1);
        d2 = fmaf(xv, w2r[d], d2);
    }
    #pragma unroll
    for (int m = 16; m >= 1; m >>= 1) {
        d1 += __shfl_xor(d1, m);
        d2 += __shfl_xor(d2, m);
    }
    if (lane == 0) {
        sel_val[tok * 2 + 0] = d1 + bias[i1];
        sel_val[tok * 2 + 1] = d2 + bias[i2];
        sel_idx[tok * 2 + 0] = i1;
        sel_idx[tok * 2 + 1] = i2;
    }
}

// ---------------------------------------------------------------------------
// Kernel 2: deterministic routing, one block per (batch, node). No atomics.
// out[b,n,:] = sum over selections routed to n of val * x[b,l,:] * gate[n,h,:]
// ---------------------------------------------------------------------------
__global__ __launch_bounds__(256)
void route(const float* __restrict__ x,
           const float* __restrict__ gates,    // (NN, NH, D)
           const float* __restrict__ sel_val,
           const int*   __restrict__ sel_idx,
           float*       __restrict__ out,      // (B, NN, D)
           float*       __restrict__ counts) { // (B, NN) as float
    const int n   = blockIdx.x;
    const int b   = blockIdx.y;
    const int tid = threadIdx.x;

    __shared__ float g[NH][D_];                // this node's 8 gate rows, 32KB
    for (int h = 0; h < NH; ++h)
        for (int d = tid; d < D_; d += 256)
            g[h][d] = gates[((size_t)n * NH + h) * D_ + d];
    __syncthreads();

    float acc[4] = {0.f, 0.f, 0.f, 0.f};
    int   cnt = 0;

    const int    base = b * L_;                // batch's first flat token
    const float* sv   = sel_val + (size_t)base * KSEL;
    const int*   si   = sel_idx + (size_t)base * KSEL;

    for (int e = 0; e < L_ * KSEL; ++e) {
        const int s = si[e];                   // uniform across block
        if ((s >> 3) == n) {
            const int    h   = s & 7;
            const float  val = sv[e];
            const int    l   = e >> 1;
            const float* xr  = x + ((size_t)base + l) * D_;
            #pragma unroll
            for (int j = 0; j < 4; ++j) {
                const int d = tid + j * 256;
                acc[j] = fmaf(val * xr[d], g[h][d], acc[j]);
            }
            ++cnt;
        }
    }

    float* orow = out + ((size_t)b * NN + n) * D_;
    #pragma unroll
    for (int j = 0; j < 4; ++j) orow[tid + j * 256] = acc[j];
    if (tid == 0) counts[b * NN + n] = (float)cnt;
}

// ---------------------------------------------------------------------------
extern "C" void kernel_launch(void* const* d_in, const int* in_sizes, int n_in,
                              void* d_out, int out_size, void* d_ws, size_t ws_size,
                              hipStream_t stream) {
    const float* x     = (const float*)d_in[0];   // (B, L, D)
    const float* W     = (const float*)d_in[1];   // (NS, D)
    const float* bias  = (const float*)d_in[2];   // (NS)
    const float* gates = (const float*)d_in[3];   // (NN, NH, D)

    char*   ws      = (char*)d_ws;
    __bf16* Wb      = (__bf16*)ws;                            // 1 MB
    float*  sel_val = (float*)(ws + (1u << 20));              // 128 KB
    int*    sel_idx = (int*)  (ws + (1u << 20) + (1u << 17)); // 128 KB

    float* out    = (float*)d_out;                 // (B, NN, D)
    float* counts = out + (size_t)B_ * NN * D_;    // (B, NN)

    cvt_w_bf16<<<(NS * D_) / (256 * 4), 256, 0, stream>>>(
        (const float4*)W, (ushort4*)Wb);

    gate_topk<<<(B_ * L_) / 16, 512, 0, stream>>>(
        x, W, Wb, bias, sel_val, sel_idx);

    dim3 grid2(NN, B_);
    route<<<grid2, 256, 0, stream>>>(x, gates, sel_val, sel_idx, out, counts);
}
